// AdaptiveAttentionLSTMCell_17789754540961
// MI455X (gfx1250) — matrix operand order repaired and checked by reading, not compiled, not run on hardware
//
#include <hip/hip_runtime.h>

// Problem constants
#define B_SZ   2048
#define DIN    2048
#define UNITS  1024
#define KTOT   (UNITS + DIN)          // 3072 combined K (h_tm cols + inputs cols)
#define NGATE  5
#define GCOLS  (NGATE * UNITS)        // 5120
#define KFEAT  49
#define BU     ((size_t)B_SZ * (size_t)UNITS)

// GEMM tiling
#define BM   128                       // rows per block (8 M-tiles)
#define BN   32                        // unit-columns per block (per gate)
#define KB   32                        // K per WMMA step (bf16 16x16x32)
#define LDA  40                        // LDS A row stride (ushort), 16B-aligned rows
#define LDBT 34                        // LDS B^T k-stride (ushort), even for u32 loads

typedef __attribute__((ext_vector_type(16))) __bf16 v16bf;
typedef __attribute__((ext_vector_type(8)))  float  v8f;

__device__ __forceinline__ unsigned short f2bf(float f) {
  union { float f; unsigned u; } v; v.f = f;
  unsigned r = v.u + 0x7FFFu + ((v.u >> 16) & 1u);   // round-to-nearest-even
  return (unsigned short)(r >> 16);
}

__device__ __forceinline__ float sigmoidf_(float x) {
  return 1.0f / (1.0f + expf(-x));
}

// ---------------- fp32 -> bf16 packing of A = [h_tm | inputs], [2048 x 3072]
__global__ __launch_bounds__(256) void pack_a(const float* __restrict__ h_tm,
                                              const float* __restrict__ x,
                                              unsigned short* __restrict__ acat) {
  int idx  = blockIdx.x * 256 + threadIdx.x;        // one bf16 pair per thread
  int col2 = (idx % (KTOT / 2)) * 2;
  int b    = idx / (KTOT / 2);
  float f0, f1;
  if (col2 < UNITS) {
    f0 = h_tm[(size_t)b * UNITS + col2];
    f1 = h_tm[(size_t)b * UNITS + col2 + 1];
  } else {
    int c = col2 - UNITS;
    f0 = x[(size_t)b * DIN + c];
    f1 = x[(size_t)b * DIN + c + 1];
  }
  unsigned v = (unsigned)f2bf(f0) | ((unsigned)f2bf(f1) << 16);
  *(unsigned*)(acat + (size_t)b * KTOT + col2) = v;
}

// ---------------- fp32 -> bf16 packing of B = [W_gates ; U_gates], [3072 x 5120]
__global__ __launch_bounds__(256) void pack_b(const float* __restrict__ W,
                                              const float* __restrict__ U,
                                              unsigned short* __restrict__ bmat) {
  long long idx = (long long)blockIdx.x * 256 + threadIdx.x;
  int col2 = (int)((idx % (GCOLS / 2)) * 2);
  int r    = (int)(idx / (GCOLS / 2));
  const float* src = (r < UNITS) ? (W + (size_t)r * GCOLS + col2)
                                 : (U + (size_t)(r - UNITS) * GCOLS + col2);
  unsigned v = (unsigned)f2bf(src[0]) | ((unsigned)f2bf(src[1]) << 16);
  *(unsigned*)(bmat + (size_t)r * GCOLS + col2) = v;
}

// ---------------- fused 5-gate GEMM + LSTM epilogue --------------------------
// Each wave: 2 M-tiles x 1 N-tile x 5 gates = 10 WMMA accumulators.
// A panel staged via CDNA5 async global->LDS DMA (ASYNCcnt), B panel staged
// K-transposed so both fragment gathers are aligned 32-bit ds loads.
// out slice0 <- st (temp), slice1 <- ht, slice2 <- mt
__global__ __launch_bounds__(256) void gate_gemm(const unsigned short* __restrict__ acat,
                                                 const unsigned short* __restrict__ bmat,
                                                 const float* __restrict__ b_gates,
                                                 const float* __restrict__ m_tm,
                                                 float* __restrict__ out) {
  __shared__ unsigned short ldsA[BM * LDA];                 // 10240 B
  __shared__ unsigned short ldsBt[NGATE * BN * LDBT];       // 10880 B

  const int tid  = threadIdx.x;
  const int lane = tid & 31;
  const int wv   = tid >> 5;          // 8 waves
  const int pair = wv >> 1;           // 0..3 : owns M-tiles {2p, 2p+1}
  const int ni   = wv & 1;            // 0..1 : 16-col strip within 32
  const int half = lane >> 4;         // 0/1
  const int l16  = lane & 15;

  const int rowbase  = blockIdx.y * BM;          // batch rows
  const int colbaseU = blockIdx.x * BN;          // unit columns
  const int colg = colbaseU + ni * 16 + l16;     // this lane's unit column

  float biasv[NGATE];
#pragma unroll
  for (int g = 0; g < NGATE; ++g) biasv[g] = b_gates[g * UNITS + colg];

  v8f acc[2][NGATE];
#pragma unroll
  for (int t = 0; t < 2; ++t)
#pragma unroll
    for (int g = 0; g < NGATE; ++g)
      acc[t][g] = (v8f){0.f,0.f,0.f,0.f,0.f,0.f,0.f,0.f};

  // A fill: 512 x 16B chunks (128 rows x 4), 2 per thread, async DMA to LDS
  const int r0 = tid >> 2,        p0 = tid & 3;
  const int r1 = (tid + 256) >> 2, p1 = (tid + 256) & 3;

  for (int kk = 0; kk < KTOT; kk += KB) {
    __syncthreads();
    // ---- stage A panel [128 x 32] bf16 via async global->LDS (no VGPR bounce)
    {
      const void* g0 = (const void*)(acat + (size_t)(rowbase + r0) * KTOT + kk + p0 * 8);
      const void* g1 = (const void*)(acat + (size_t)(rowbase + r1) * KTOT + kk + p1 * 8);
      unsigned l0 = (unsigned)(uintptr_t)(ldsA + r0 * LDA + p0 * 8);
      unsigned l1 = (unsigned)(uintptr_t)(ldsA + r1 * LDA + p1 * 8);
      asm volatile("global_load_async_to_lds_b128 %0, %1, off"
                   :: "v"(l0), "v"(g0) : "memory");
      asm volatile("global_load_async_to_lds_b128 %0, %1, off"
                   :: "v"(l1), "v"(g1) : "memory");
      if (kk + KB < KTOT)
        __builtin_prefetch(acat + (size_t)(rowbase + r0) * KTOT + (kk + KB) + p0 * 8, 0, 1);
    }
    // ---- stage B panel, K-transposed: ldsBt[g][n][k], 5 x (32n x 32k)
    for (int i = tid; i < NGATE * 32 * 8; i += 256) {
      int seg  = i >> 3;               // (g,k) pair: 160 segments
      int part = i & 7;                // 4-column chunk
      int g = seg >> 5, k = seg & 31;
      const uint2* gp = (const uint2*)(bmat + (size_t)(kk + k) * GCOLS + g * UNITS + colbaseU + part * 4);
      uint2 v = *gp;
      unsigned short* bp = ldsBt + g * (BN * LDBT) + (part * 4) * LDBT + k;
      bp[0 * LDBT] = (unsigned short)(v.x & 0xFFFFu);
      bp[1 * LDBT] = (unsigned short)(v.x >> 16);
      bp[2 * LDBT] = (unsigned short)(v.y & 0xFFFFu);
      bp[3 * LDBT] = (unsigned short)(v.y >> 16);
      if (kk + KB < KTOT)
        __builtin_prefetch(bmat + (size_t)(kk + KB + k) * GCOLS + g * UNITS + colbaseU + part * 4, 0, 1);
    }
    asm volatile("s_wait_asynccnt 0x0" ::: "memory");
    __syncthreads();

    // ---- A fragments (2 M-tiles): 16x32 bf16 ISA layout
    union U { v16bf v; unsigned u[8]; };
    U af0, af1, bfr[NGATE];
    {
      const unsigned short* ap0 = ldsA + (pair * 32 + l16) * LDA + 8 * half;
      const unsigned short* ap1 = ap0 + 16 * LDA;
#pragma unroll
      for (int j = 0; j < 4; ++j) {
        af0.u[j]     = *(const unsigned*)(ap0 + 2 * j);
        af0.u[j + 4] = *(const unsigned*)(ap0 + 16 + 2 * j);
        af1.u[j]     = *(const unsigned*)(ap1 + 2 * j);
        af1.u[j + 4] = *(const unsigned*)(ap1 + 16 + 2 * j);
      }
    }
    // ---- preload all 5 B fragments, then issue 10 WMMAs back-to-back
#pragma unroll
    for (int g = 0; g < NGATE; ++g) {
      const unsigned short* bp = ldsBt + g * (BN * LDBT) + (ni * 16 + l16) * LDBT + 8 * half;
#pragma unroll
      for (int j = 0; j < 8; ++j) bfr[g].u[j] = *(const unsigned*)(bp + 2 * j + (j >= 4 ? 8 : 0));
    }
#pragma unroll
    for (int g = 0; g < NGATE; ++g)
      acc[0][g] = __builtin_amdgcn_wmma_f32_16x16x32_bf16(
          false, af0.v, false, bfr[g].v, (short)0, acc[0][g], false, false);
#pragma unroll
    for (int g = 0; g < NGATE; ++g)
      acc[1][g] = __builtin_amdgcn_wmma_f32_16x16x32_bf16(
          false, af1.v, false, bfr[g].v, (short)0, acc[1][g], false, false);
  }

  // ---- fused LSTM epilogue. C layout: VGPR r -> row r (lanes<16) / r+8 (lanes>=16)
#pragma unroll
  for (int t = 0; t < 2; ++t) {
#pragma unroll
    for (int r = 0; r < 8; ++r) {
      int row = rowbase + (pair * 2 + t) * 16 + r + 8 * half;
      float zf = acc[t][0][r] + biasv[0];
      float zi = acc[t][1][r] + biasv[1];
      float zo = acc[t][2][r] + biasv[2];
      float zg = acc[t][3][r] + biasv[3];
      float za = acc[t][4][r] + biasv[4];
      float ft = sigmoidf_(zf);
      float it = sigmoidf_(zi);
      float ot = sigmoidf_(zo);
      float gt = sigmoidf_(zg);
      float at = tanhf(za);
      float mtm = m_tm[(size_t)row * UNITS + colg];
      float mt  = mtm * ft + it * at;
      float tm  = tanhf(mt);
      float ht  = ot * tm;
      float st  = gt * tm;
      size_t o  = (size_t)row * UNITS + colg;
      out[o]          = st;   // temp in slice0, consumed by finalize
      out[BU + o]     = ht;
      out[2 * BU + o] = mt;
    }
  }
}

// ---------------- finalize: out0 = ht + st + sum_k v_seq  (softmax over size-1
// axis == 1, so ct = sum_k v_expand). Pure bandwidth kernel: 411 MB of v_seq.
__global__ __launch_bounds__(256) void finalize(const float* __restrict__ v_seq,
                                                float* __restrict__ out) {
  int idx = blockIdx.x * 256 + threadIdx.x;     // BU/4 threads, float4 each
  int b   = idx >> 8;                           // 256 float4 per row of 1024
  int u4  = idx & 255;
  const float4* vp = (const float4*)v_seq + (size_t)b * KFEAT * 256 + u4;
  float4 acc = {0.f, 0.f, 0.f, 0.f};
#pragma unroll
  for (int k = 0; k < KFEAT; ++k) {
    float4 t = vp[(size_t)k * 256];
    acc.x += t.x; acc.y += t.y; acc.z += t.z; acc.w += t.w;
  }
  size_t o = (size_t)b * 256 + u4;
  float4 s = ((const float4*)out)[o];                 // st (temp)
  float4 h = ((const float4*)(out + BU))[o];          // ht
  float4 rr;
  rr.x = acc.x + s.x + h.x;
  rr.y = acc.y + s.y + h.y;
  rr.z = acc.z + s.z + h.z;
  rr.w = acc.w + s.w + h.w;
  ((float4*)out)[o] = rr;
}

extern "C" void kernel_launch(void* const* d_in, const int* in_sizes, int n_in,
                              void* d_out, int out_size, void* d_ws, size_t ws_size,
                              hipStream_t stream) {
  (void)in_sizes; (void)n_in; (void)out_size; (void)ws_size;
  const float* inputs = (const float*)d_in[0];
  const float* h_tm   = (const float*)d_in[1];
  const float* m_tm   = (const float*)d_in[2];
  const float* v_seq  = (const float*)d_in[3];
  const float* W_g    = (const float*)d_in[4];
  const float* U_g    = (const float*)d_in[5];
  const float* b_g    = (const float*)d_in[6];
  // d_in[7..9] (W_z, U_z, W_h) are dead: softmax over a size-1 axis is 1.
  float* out = (float*)d_out;

  unsigned short* acat = (unsigned short*)d_ws;                 // [2048 x 3072] bf16
  unsigned short* bmat = acat + (size_t)B_SZ * KTOT;            // [3072 x 5120] bf16
  // total ws use: (2048*3072 + 3072*5120)*2 = 42 MiB

  pack_a<<<(B_SZ * KTOT / 2) / 256, 256, 0, stream>>>(h_tm, inputs, acat);
  pack_b<<<(KTOT * GCOLS / 2) / 256, 256, 0, stream>>>(W_g, U_g, bmat);

  dim3 grid(UNITS / BN, B_SZ / BM);   // (32, 16)
  gate_gemm<<<grid, 256, 0, stream>>>(acat, bmat, b_g, m_tm, out);

  finalize<<<(unsigned)(BU / 4 / 256), 256, 0, stream>>>(v_seq, out);
}
